// MambaGenerator_63024350101532
// MI455X (gfx1250) — matrix operand order repaired and checked by reading, not compiled, not run on hardware
//
#include <hip/hip_runtime.h>
#include <hip/hip_bf16.h>

#define BSZ    16
#define LSEQ   256
#define DMODEL 256
#define DINNER 512
#define DSTATE 16
#define DTRANK 16
#define TLEN   16
#define ROWS   (BSZ * LSEQ)      // 4096
#define KPAD   32                // padded K for the delta GEMM

typedef __attribute__((ext_vector_type(16))) __bf16        v16bf;
typedef __attribute__((ext_vector_type(8)))  float         v8f;
typedef __attribute__((ext_vector_type(8)))  unsigned int  v8u;
typedef __attribute__((ext_vector_type(4)))  unsigned int  u32x4;
typedef __attribute__((ext_vector_type(2)))  unsigned int  u32x2;
typedef __attribute__((ext_vector_type(8)))  int           i32x8;
typedef __attribute__((ext_vector_type(4)))  int           i32x4;

__device__ __forceinline__ unsigned int f32_to_bf16_bits(float f) {
    unsigned int u = __float_as_uint(f);
    u += 0x7FFFu + ((u >> 16) & 1u);   // round-to-nearest-even
    return u >> 16;
}

// ---------------------------------------------------------------------------
// C[M x N] = A[M x K](f32, row stride lda) * W[N x K]^T (bf16 bits, row-major)
// Requires K % 32 == 0, N % 16 == 0.
// Block = 256 threads = 8 waves; block tile 32(M) x 128(N); wave tile 32x16
// (two 16x16 WMMAs sharing one B fragment). A slab (32x32) staged fp32->bf16
// into double-buffered LDS. Wave 0 additionally TDM-prefetches the block's
// first 32x128 B panel into scratch LDS (tensor_load_to_lds + tensorcnt wait).
// ---------------------------------------------------------------------------
__global__ __launch_bounds__(256)
void gemm_wmma_bf16(const float* __restrict__ A, int lda,
                    const unsigned short* __restrict__ Wb,
                    float* __restrict__ C, int ldc,
                    int M, int N, int K)
{
    __shared__ unsigned int Alds[2][32][20];   // 32 rows x 16 bf16-pairs, +4 pad (16B align kept)
    __shared__ unsigned int Bpf[2048];         // 8KB scratch for TDM panel prefetch

    const int tid  = threadIdx.x;
    const int wave = tid >> 5;
    const int lane = tid & 31;
    const int ml   = lane & 15;   // fragment row (A) / col (B,C)
    const int kh   = lane >> 4;   // K-half selector

    const int mBase  = blockIdx.y * 32;
    const int nPanel = blockIdx.x * 128;
    const int nBase  = nPanel + wave * 16;
    const int n      = nBase + ml;
    const int ncl    = (n < N) ? n : (N - 1);          // clamp for safe loads

    // ---- TDM: async-tensor prefetch of the first B panel (32 K x 128 N) ----
    if (wave == 0) {
        unsigned long long ga = (unsigned long long)(const void*)(Wb + (size_t)nPanel * K);
        unsigned int ldsoff   = (unsigned int)(unsigned long long)(uintptr_t)&Bpf[0];
        int remRows = N - nPanel; if (remRows > 128) remRows = 128;
        u32x4 g0;
        g0[0] = 1u;                                        // count=1 (valid user D#)
        g0[1] = ldsoff;                                    // lds_addr
        g0[2] = (unsigned int)(ga & 0xFFFFFFFFu);          // global_addr[31:0]
        g0[3] = (unsigned int)((ga >> 32) & 0x01FFFFFFu)   // global_addr[56:32]
              | (2u << 30);                                // type=2 ("image")
        i32x8 g1;
        g1[0] = (1 << 16);                                 // data_size=1 (2 bytes)
        g1[1] = (K & 0xFFFF) << 16;                        // tensor_dim0[15:0]
        g1[2] = ((K >> 16) & 0xFFFF) | ((remRows & 0xFFFF) << 16); // dim0 hi | dim1 lo
        g1[3] = (32 << 16);                                // dim1 hi(=0) | tile_dim0=32
        g1[4] = remRows & 0xFFFF;                          // tile_dim1 | tile_dim2=0
        g1[5] = K;                                         // tensor_dim0_stride[31:0]
        g1[6] = 0;                                         // stride hi | dim1_stride lo
        g1[7] = 0;
        i32x4 gz4 = {};
        i32x8 gz8 = {};
        __builtin_amdgcn_tensor_load_to_lds(g0, g1, gz4, gz4, gz8, 0);
        __builtin_amdgcn_s_wait_tensorcnt(0);
    }

    v8f acc0 = {};
    v8f acc1 = {};
    const unsigned short* wrow = Wb + (size_t)ncl * K;

    // staging coords: each thread covers 4 consecutive floats (2 bf16 pairs)
    const int srow = tid >> 3;                 // 0..31
    const int scg  = tid & 7;                  // group of 4 cols
    const int srcl = (mBase + srow < M) ? (mBase + srow) : (M - 1);  // clamp rows
    const float* arow = A + (size_t)srcl * lda + 4 * scg;
    int buf = 0;

    for (int k0 = 0; k0 < K; k0 += 32) {
        // ---- stage A 32x32 slab: one float4 load + one packed b64 LDS store ----
        {
            float4 f = *(const float4*)(arow + k0);
            u32x2 p;
            p[0] = f32_to_bf16_bits(f.x) | (f32_to_bf16_bits(f.y) << 16);
            p[1] = f32_to_bf16_bits(f.z) | (f32_to_bf16_bits(f.w) << 16);
            *(u32x2*)&Alds[buf][srow][scg * 2] = p;
        }
        __syncthreads();

        // ---- A fragments for both 16-row subtiles: 16B LDS reads ----
        u32x4 a00 = *(const u32x4*)&Alds[buf][ml][kh * 4];
        u32x4 a01 = *(const u32x4*)&Alds[buf][ml][kh * 4 + 8];
        u32x4 a10 = *(const u32x4*)&Alds[buf][16 + ml][kh * 4];
        u32x4 a11 = *(const u32x4*)&Alds[buf][16 + ml][kh * 4 + 8];
        v8u au0 = __builtin_shufflevector(a00, a01, 0, 1, 2, 3, 4, 5, 6, 7);
        v8u au1 = __builtin_shufflevector(a10, a11, 0, 1, 2, 3, 4, 5, 6, 7);

        // ---- B fragment: two 16B global reads (col ncl, k = 8kh..+7 and 16+8kh..+7) ----
        const unsigned short* bp = wrow + k0 + kh * 8;
        u32x4 b0 = *(const u32x4*)bp;
        u32x4 b1 = *(const u32x4*)(bp + 16);
        v8u bu = __builtin_shufflevector(b0, b1, 0, 1, 2, 3, 4, 5, 6, 7);

        if (k0 + 32 < K)
            __builtin_prefetch(wrow + k0 + 32, 0, 1);      // global_prefetch_b8

        v16bf bfr = __builtin_bit_cast(v16bf, bu);
        acc0 = __builtin_amdgcn_wmma_f32_16x16x32_bf16(
                   false, __builtin_bit_cast(v16bf, au0), false, bfr,
                   (short)0, acc0, false, false);
        acc1 = __builtin_amdgcn_wmma_f32_16x16x32_bf16(
                   false, __builtin_bit_cast(v16bf, au1), false, bfr,
                   (short)0, acc1, false, false);
        buf ^= 1;                                          // double buffer: no 2nd barrier
    }

    // ---- store C tiles: lane ml = col; rows kh*8 + j (+16 for second tile) ----
    if (n < N) {
        #pragma unroll
        for (int j = 0; j < 8; ++j) {
            int m0 = mBase + kh * 8 + j;
            int m1 = m0 + 16;
            if (m0 < M) C[(size_t)m0 * ldc + n] = acc0[j];
            if (m1 < M) C[(size_t)m1 * ldc + n] = acc1[j];
        }
    }
}

// ---------------------------------------------------------------------------
// elementwise / setup kernels
// ---------------------------------------------------------------------------
__global__ void transpose_x_kernel(const float* __restrict__ x, float* __restrict__ inp) {
    int idx = blockIdx.x * blockDim.x + threadIdx.x;           // [B][L][D]
    if (idx >= BSZ * LSEQ * DMODEL) return;
    int d = idx & (DMODEL - 1);
    int l = (idx >> 8) & (LSEQ - 1);
    int b = idx >> 16;
    inp[idx] = x[((long)b * DMODEL + d) * LSEQ + l];           // x: [B][D][L]
}

__global__ void cvt_bf16_kernel(const float* __restrict__ s, unsigned short* __restrict__ d, int n) {
    int i = blockIdx.x * blockDim.x + threadIdx.x;
    if (i < n) d[i] = (unsigned short)f32_to_bf16_bits(s[i]);
}

// W_dt [512x16] f32 -> bf16 zero-padded to [512x32]
__global__ void cvt_wdt_pad_kernel(const float* __restrict__ s, unsigned short* __restrict__ d) {
    int i = blockIdx.x * blockDim.x + threadIdx.x;
    if (i >= DINNER * KPAD) return;
    int c = i & (KPAD - 1), r = i >> 5;
    d[i] = (c < DTRANK) ? (unsigned short)f32_to_bf16_bits(s[r * DTRANK + c]) : (unsigned short)0;
}

// dt = x_dbl[:, :16] -> zero-padded [ROWS x 32]
__global__ void pack_dt_kernel(const float* __restrict__ xdbl, float* __restrict__ dtp) {
    int i = blockIdx.x * blockDim.x + threadIdx.x;
    if (i >= ROWS * KPAD) return;
    int c = i & (KPAD - 1), r = i >> 5;
    dtp[i] = (c < DTRANK) ? xdbl[r * 48 + c] : 0.0f;
}

__global__ void neg_exp_kernel(const float* __restrict__ a, float* __restrict__ o, int n) {
    int i = blockIdx.x * blockDim.x + threadIdx.x;
    if (i < n) o[i] = -__expf(a[i]);
}

// causal depthwise conv1d (width 4) + bias + SiLU; input = first half of xz
__global__ void conv_silu_kernel(const float* __restrict__ xz,
                                 const float* __restrict__ cw,
                                 const float* __restrict__ cb,
                                 float* __restrict__ xi) {
    int idx = blockIdx.x * blockDim.x + threadIdx.x;           // [B][L][DINNER]
    if (idx >= ROWS * DINNER) return;
    int c  = idx & (DINNER - 1);
    int bl = idx >> 9;
    int l  = bl & (LSEQ - 1);
    int b  = bl >> 8;
    float acc = cb[c];
    #pragma unroll
    for (int j = 0; j < 4; ++j) {
        int ls = l - 3 + j;
        if (ls >= 0) acc += cw[c * 4 + j] * xz[((long)((b << 8) + ls)) * (2 * DINNER) + c];
    }
    xi[idx] = acc / (1.0f + __expf(-acc));                     // SiLU
}

__global__ void bias_softplus_kernel(float* __restrict__ delta, const float* __restrict__ b_dt) {
    int i = blockIdx.x * blockDim.x + threadIdx.x;
    if (i >= ROWS * DINNER) return;
    float v = delta[i] + b_dt[i & (DINNER - 1)];
    delta[i] = fmaxf(v, 0.0f) + log1pf(__expf(-fabsf(v)));     // softplus
}

// selective scan: one lane per (b, channel); h[16] in registers; fused *silu(z)
__global__ __launch_bounds__(256)
void scan_kernel(const float* __restrict__ xi,   const float* __restrict__ delta,
                 const float* __restrict__ xdbl, const float* __restrict__ Aneg,
                 const float* __restrict__ Dp,   const float* __restrict__ xz,
                 float* __restrict__ y) {
    int b = blockIdx.x >> 1;
    int d = ((blockIdx.x & 1) << 8) + threadIdx.x;             // channel 0..511
    float h[DSTATE];
    float Ad[DSTATE];
    #pragma unroll
    for (int nn = 0; nn < DSTATE; ++nn) { h[nn] = 0.0f; Ad[nn] = Aneg[d * DSTATE + nn]; }
    const float Dv = Dp[d];
    for (int l = 0; l < LSEQ; ++l) {
        long row = (long)(b * LSEQ + l);
        float dl = delta[row * DINNER + d];
        float u  = xi[row * DINNER + d];
        const float* Bm = xdbl + row * 48 + DTRANK;            // cols 16..31
        const float* Cm = xdbl + row * 48 + DTRANK + DSTATE;   // cols 32..47
        float du = dl * u, acc = 0.0f;
        #pragma unroll
        for (int nn = 0; nn < DSTATE; ++nn) {
            float dA = __expf(dl * Ad[nn]);
            h[nn] = dA * h[nn] + du * Bm[nn];
            acc  += h[nn] * Cm[nn];
        }
        float z  = xz[row * (2 * DINNER) + DINNER + d];
        float sz = z / (1.0f + __expf(-z));
        y[row * DINNER + d] = (acc + u * Dv) * sz;
    }
}

// shift window (ping-pong) + append new token + emit output column t
__global__ void update_window_kernel(const float* __restrict__ inpCur,
                                     const float* __restrict__ o,
                                     float* __restrict__ inpNext,
                                     float* __restrict__ out, int t) {
    int idx = blockIdx.x * blockDim.x + threadIdx.x;           // [B][L][D]
    if (idx >= BSZ * LSEQ * DMODEL) return;
    int d = idx & (DMODEL - 1);
    int l = (idx >> 8) & (LSEQ - 1);
    int b = idx >> 16;
    float v;
    if (l < LSEQ - 1) v = inpCur[(((long)(b << 8)) + l + 1) * DMODEL + d];
    else {
        v = o[b * DMODEL + d];
        out[((long)b * DMODEL + d) * TLEN + t] = v;            // out: [B][D][T]
    }
    inpNext[idx] = v;
}

// ---------------------------------------------------------------------------
extern "C" void kernel_launch(void* const* d_in, const int* in_sizes, int n_in,
                              void* d_out, int out_size, void* d_ws, size_t ws_size,
                              hipStream_t stream) {
    const float* x      = (const float*)d_in[0];
    const float* W_in   = (const float*)d_in[1];
    const float* conv_w = (const float*)d_in[2];
    const float* conv_b = (const float*)d_in[3];
    const float* W_x    = (const float*)d_in[4];
    const float* W_dt   = (const float*)d_in[5];
    const float* b_dt   = (const float*)d_in[6];
    const float* A_log  = (const float*)d_in[7];
    const float* Dp     = (const float*)d_in[8];
    const float* W_out  = (const float*)d_in[9];
    float* out = (float*)d_out;
    (void)in_sizes; (void)n_in; (void)out_size; (void)ws_size;

    unsigned char* ws = (unsigned char*)d_ws;
    size_t off = 0;
    auto carve = [&](size_t bytes) -> unsigned char* {
        unsigned char* p = ws + off;
        off = (off + bytes + 255) & ~(size_t)255;
        return p;
    };
    float* inpA  = (float*)carve((size_t)BSZ * LSEQ * DMODEL * 4);
    float* inpB  = (float*)carve((size_t)BSZ * LSEQ * DMODEL * 4);
    float* xz    = (float*)carve((size_t)ROWS * 2 * DINNER * 4);
    float* xi    = (float*)carve((size_t)ROWS * DINNER * 4);
    float* xdbl  = (float*)carve((size_t)ROWS * 48 * 4);
    float* dtpad = (float*)carve((size_t)ROWS * KPAD * 4);
    float* delta = (float*)carve((size_t)ROWS * DINNER * 4);
    float* ybuf  = (float*)carve((size_t)ROWS * DINNER * 4);
    float* obuf  = (float*)carve((size_t)BSZ * DMODEL * 4);
    unsigned short* Win_bf  = (unsigned short*)carve((size_t)2 * DINNER * DMODEL * 2);
    unsigned short* Wx_bf   = (unsigned short*)carve((size_t)48 * DINNER * 2);
    unsigned short* Wdt_bf  = (unsigned short*)carve((size_t)DINNER * KPAD * 2);
    unsigned short* Wout_bf = (unsigned short*)carve((size_t)DMODEL * DINNER * 2);
    float* Aneg  = (float*)carve((size_t)DINNER * DSTATE * 4);

    auto g1 = [](int n) { return (n + 255) / 256; };

    // setup (deterministic every call)
    transpose_x_kernel<<<g1(BSZ * LSEQ * DMODEL), 256, 0, stream>>>(x, inpA);
    cvt_bf16_kernel<<<g1(2 * DINNER * DMODEL), 256, 0, stream>>>(W_in, Win_bf, 2 * DINNER * DMODEL);
    cvt_bf16_kernel<<<g1(48 * DINNER), 256, 0, stream>>>(W_x, Wx_bf, 48 * DINNER);
    cvt_wdt_pad_kernel<<<g1(DINNER * KPAD), 256, 0, stream>>>(W_dt, Wdt_bf);
    cvt_bf16_kernel<<<g1(DMODEL * DINNER), 256, 0, stream>>>(W_out, Wout_bf, DMODEL * DINNER);
    neg_exp_kernel<<<g1(DINNER * DSTATE), 256, 0, stream>>>(A_log, Aneg, DINNER * DSTATE);

    for (int t = 0; t < TLEN; ++t) {
        const float* inpCur = (t & 1) ? inpB : inpA;
        float*       inpNext = (t & 1) ? inpA : inpB;

        // xz = inp @ W_in^T            [4096 x 1024], K=256
        gemm_wmma_bf16<<<dim3(8, ROWS / 32), 256, 0, stream>>>(
            inpCur, DMODEL, Win_bf, xz, 2 * DINNER, ROWS, 2 * DINNER, DMODEL);

        // causal depthwise conv + SiLU -> xi
        conv_silu_kernel<<<g1(ROWS * DINNER), 256, 0, stream>>>(xz, conv_w, conv_b, xi);

        // x_dbl = xi @ W_x^T           [4096 x 48], K=512
        gemm_wmma_bf16<<<dim3(1, ROWS / 32), 256, 0, stream>>>(
            xi, DINNER, Wx_bf, xdbl, 48, ROWS, 48, DINNER);

        // delta_pre = dt @ W_dt^T      [4096 x 512], K=32 (zero-padded)
        pack_dt_kernel<<<g1(ROWS * KPAD), 256, 0, stream>>>(xdbl, dtpad);
        gemm_wmma_bf16<<<dim3(4, ROWS / 32), 256, 0, stream>>>(
            dtpad, KPAD, Wdt_bf, delta, DINNER, ROWS, DINNER, KPAD);
        bias_softplus_kernel<<<g1(ROWS * DINNER), 256, 0, stream>>>(delta, b_dt);

        // selective scan (fused + u*Dp and *silu(z))
        scan_kernel<<<32, 256, 0, stream>>>(xi, delta, xdbl, Aneg, Dp, xz, ybuf);

        // only last position per batch: o = y[:, L-1, :] @ W_out^T  [16 x 256], K=512
        gemm_wmma_bf16<<<dim3(2, 1), 256, 0, stream>>>(
            ybuf + (size_t)(LSEQ - 1) * DINNER, LSEQ * DINNER,
            Wout_bf, obuf, DMODEL, BSZ, DMODEL, DINNER);

        // shift window, append token, emit output column
        update_window_kernel<<<g1(BSZ * LSEQ * DMODEL), 256, 0, stream>>>(
            inpCur, obuf, inpNext, out, t);
    }
}
